// WIAA_75582834475530
// MI455X (gfx1250) — compile-verified, matrix operand order checked by reading, not run on hardware
//
#include <hip/hip_runtime.h>
#include <hip/hip_bf16.h>

// ---------------------------------------------------------------------------
// Shapes (fixed by the reference): B=32, C=N=1024, K(haar)=512.
// Workspace layout (bf16 elements), ~226 MB total:
//   A   [32768,512]   haar coeffs
//   Wqb/Wkb/Wvb [1024,512] packed weights (first 512 cols only — rest hit zeros)
//   Q,K [32,1024,1024], VT [32,1024(d),1024(c)]  (V stored transposed)
// ---------------------------------------------------------------------------

typedef __attribute__((ext_vector_type(16))) __bf16        v16bf;
typedef __attribute__((ext_vector_type(8)))  float         v8f;
typedef __attribute__((ext_vector_type(4)))  unsigned int  v4u;

union FragU { v4u u[2]; v16bf v; };

// 16-bit A-matrix 16x32 fragment per ISA layout:
//   lane L -> row L%16 ; lanes 0-15 hold K {0..7,16..23}, lanes 16-31 {8..15,24..31}
// B is loaded with the identical pattern from the N-major (N x K) operand,
// i.e. lane L -> column L%16 of B == row L%16 of the N x K array.
__device__ __forceinline__ v16bf load_frag(const __bf16* tile, int ld, int lane) {
    const int r  = lane & 15;
    const int k0 = (lane < 16) ? 0 : 8;
    const __bf16* p = tile + (size_t)r * ld + k0;
    FragU f;
    f.u[0] = *(const v4u*)(p);        // K chunk [k0, k0+8)
    f.u[1] = *(const v4u*)(p + 16);   // K chunk [k0+16, k0+24)
    return f.v;
}

__device__ __forceinline__ v8f wmma_bf16(v16bf a, v16bf b, v8f c) {
    return __builtin_amdgcn_wmma_f32_16x16x32_bf16(
               false, a, false, b, (short)0, c, false, false);
}

// ---- Kernel 1: haar level-1 approx, f32 -> bf16 ---------------------------
__global__ void wiaa_haar_bf16(const float* __restrict__ x, __bf16* __restrict__ A) {
    const size_t i   = (size_t)blockIdx.x * blockDim.x + threadIdx.x;  // 32768*512
    const size_t row = i >> 9;
    const size_t k   = i & 511;
    const float* xr  = x + (row << 10) + (k << 1);
    A[i] = (__bf16)((xr[0] + xr[1]) * 0.70710678118654752f);
}

// ---- Kernel 2: pack W[:, :512] to bf16 [1024,512] -------------------------
__global__ void wiaa_wpack_bf16(const float* __restrict__ W, __bf16* __restrict__ Wb) {
    const size_t i = (size_t)blockIdx.x * blockDim.x + threadIdx.x;    // 1024*512
    const size_t d = i >> 9;
    const size_t k = i & 511;
    Wb[i] = (__bf16)W[(d << 10) + k];
}

// ---- Kernel 3: GEMM  out[M=32768, N=1024] = A[M,512] * W[N,512]^T + bias --
// One wave per 32x32 macro-tile (2x2 register-blocked WMMA tiles):
// per K-step 4 fragment loads feed 4 WMMAs -> 2x the FLOP/byte of a 16x16
// tiling, keeping the XDL pipe fed from L2.
// transposed==1 (V path): store out[b, d, c] instead of out[bc, d].
__global__ __launch_bounds__(256) void wiaa_gemm_bf16(
        const __bf16* __restrict__ A, const __bf16* __restrict__ W,
        const float* __restrict__ bias, __bf16* __restrict__ out, int transposed) {
    const int lane = threadIdx.x & 31;
    const int wave = threadIdx.x >> 5;
    const int tile = blockIdx.x * 8 + wave;     // 32768 tiles = 1024 mt x 32 nt
    const int mt = tile >> 5;
    const int nt = tile & 31;
    const int mrow = mt * 32;
    const int ncol = nt * 32;

    const __bf16* A0 = A + (size_t)mrow * 512;
    const __bf16* A1 = A0 + (size_t)16 * 512;
    const __bf16* W0 = W + (size_t)ncol * 512;
    const __bf16* W1 = W0 + (size_t)16 * 512;

    v8f acc00 = {}, acc01 = {}, acc10 = {}, acc11 = {};
    for (int k0 = 0; k0 < 512; k0 += 32) {
        __builtin_prefetch(A0 + k0 + 128, 0, 1);   // global_prefetch_b8
        __builtin_prefetch(W0 + k0 + 128, 0, 1);
        v16bf a0 = load_frag(A0 + k0, 512, lane);
        v16bf a1 = load_frag(A1 + k0, 512, lane);
        v16bf b0 = load_frag(W0 + k0, 512, lane);
        v16bf b1 = load_frag(W1 + k0, 512, lane);
        acc00 = wmma_bf16(a0, b0, acc00);
        acc01 = wmma_bf16(a0, b1, acc01);
        acc10 = wmma_bf16(a1, b0, acc10);
        acc11 = wmma_bf16(a1, b1, acc11);
    }

    const int crow = (lane >> 4) << 3;   // C/D layout: vgpr j -> M=j (+8 for hi lanes)
    const int ccol = lane & 15;
    const float bv0 = bias[ncol + ccol];
    const float bv1 = bias[ncol + 16 + ccol];

#pragma unroll
    for (int mi = 0; mi < 2; ++mi) {
        const v8f* accs[2] = { mi ? &acc10 : &acc00, mi ? &acc11 : &acc01 };
        if (!transposed) {
#pragma unroll
            for (int j = 0; j < 8; ++j) {
                const int r = mrow + mi * 16 + crow + j;
                out[(size_t)r * 1024 + ncol + ccol]      = (__bf16)((*accs[0])[j] + bv0);
                out[(size_t)r * 1024 + ncol + 16 + ccol] = (__bf16)((*accs[1])[j] + bv1);
            }
        } else {
#pragma unroll
            for (int j = 0; j < 8; ++j) {
                const int r = mrow + mi * 16 + crow + j;   // r = b*1024 + c
                const int b = r >> 10, c = r & 1023;
                out[((size_t)b * 1024 + (ncol + ccol)) * 1024 + c] =
                    (__bf16)((*accs[0])[j] + bv0);
                out[((size_t)b * 1024 + (ncol + 16 + ccol)) * 1024 + c] =
                    (__bf16)((*accs[1])[j] + bv1);
            }
        }
    }
}

// ---- Kernel 4: attention for one (batch, 16-row query tile) ---------------
// Phase 1: S(16x1024) = q_tile * k^T * (1/32)  (WMMA, 1x4 blocked: one q-frag
//          reused across 4 key tiles per K-step)
// Phase 2: row-wise softmax in LDS (f32 math, bf16 storage, in-place)
// Phase 3: O = P(16x1024) @ V(1024x1024), 1x4 blocked (P-frag from LDS reused
//          across 4 V-tiles); fused final transpose + residual store.
__global__ __launch_bounds__(256) void wiaa_attn(
        const __bf16* __restrict__ Q, const __bf16* __restrict__ Kb,
        const __bf16* __restrict__ VT, const float* __restrict__ x,
        float* __restrict__ out) {
    __shared__ __align__(16) __bf16 s_sc[16 * 1024];   // 32 KB score/prob strip
    __shared__ float s_red[16 * 16];
    __shared__ float s_rowv[16];

    const int b    = blockIdx.x >> 6;
    const int mt   = blockIdx.x & 63;
    const int mrow = mt * 16;
    const int lane = threadIdx.x & 31;
    const int wave = threadIdx.x >> 5;
    const int crow = (lane >> 4) << 3;
    const int ccol = lane & 15;

    const __bf16* qb = Q  + ((size_t)b << 20) + (size_t)mrow * 1024;
    const __bf16* kb = Kb + ((size_t)b << 20);
    const __bf16* vb = VT + ((size_t)b << 20);

    // Phase 1: scores (each wave owns 8 key tiles, processed 4 at a time)
    for (int io = 0; io < 2; ++io) {
        const int et0 = wave * 8 + io * 4;
        const __bf16* kt = kb + (size_t)et0 * 16 * 1024;
        v8f acc[4] = {};
        for (int k0 = 0; k0 < 1024; k0 += 32) {
            v16bf a = load_frag(qb + k0, 1024, lane);
#pragma unroll
            for (int t = 0; t < 4; ++t) {
                v16bf bf = load_frag(kt + (size_t)t * 16 * 1024 + k0, 1024, lane);
                acc[t] = wmma_bf16(a, bf, acc[t]);
            }
        }
#pragma unroll
        for (int t = 0; t < 4; ++t)
#pragma unroll
            for (int j = 0; j < 8; ++j)
                s_sc[(crow + j) * 1024 + (et0 + t) * 16 + ccol] =
                    (__bf16)(acc[t][j] * 0.03125f);
    }
    __syncthreads();

    // Phase 2: softmax; thread owns (row = tid&15, 64-elem chunk = tid>>4)
    {
        const int row = threadIdx.x & 15;
        const int chunk = threadIdx.x >> 4;
        __bf16* rp = s_sc + row * 1024 + chunk * 64;
        float m = -3.4e38f;
        for (int t = 0; t < 64; ++t) m = fmaxf(m, (float)rp[t]);
        s_red[row * 16 + chunk] = m;
        __syncthreads();
        if (threadIdx.x < 16) {
            float mm = s_red[threadIdx.x * 16];
            for (int t = 1; t < 16; ++t) mm = fmaxf(mm, s_red[threadIdx.x * 16 + t]);
            s_rowv[threadIdx.x] = mm;
        }
        __syncthreads();
        const float rmax = s_rowv[row];
        float ssum = 0.f;
        for (int t = 0; t < 64; ++t) {
            const float e = __expf((float)rp[t] - rmax);
            ssum += e;
            rp[t] = (__bf16)e;                 // in-place, own elements only
        }
        __syncthreads();
        s_red[row * 16 + chunk] = ssum;
        __syncthreads();
        if (threadIdx.x < 16) {
            float ss = 0.f;
            for (int t = 0; t < 16; ++t) ss += s_red[threadIdx.x * 16 + t];
            s_rowv[threadIdx.x] = ss;
        }
        __syncthreads();
        const float inv = 1.0f / s_rowv[row];
        for (int t = 0; t < 64; ++t) rp[t] = (__bf16)((float)rp[t] * inv);
    }
    __syncthreads();

    // Phase 3: out = P @ V, 4 n-tiles per pass; fused transpose + residual
    for (int io = 0; io < 2; ++io) {
        const int nt0 = wave * 8 + io * 4;
        const __bf16* vt = vb + (size_t)nt0 * 16 * 1024;
        v8f acc[4] = {};
        for (int e0 = 0; e0 < 1024; e0 += 32) {
            v16bf a = load_frag(s_sc + e0, 1024, lane);    // P rows from LDS
#pragma unroll
            for (int t = 0; t < 4; ++t) {
                v16bf bf = load_frag(vt + (size_t)t * 16 * 1024 + e0, 1024, lane);
                acc[t] = wmma_bf16(a, bf, acc[t]);
            }
        }
#pragma unroll
        for (int t = 0; t < 4; ++t) {
            const int n = (nt0 + t) * 16 + ccol;
#pragma unroll
            for (int j = 0; j < 8; ++j) {
                const int c = mrow + crow + j;
                const size_t idx = (((size_t)b * 1024) + n) * 1024 + c;
                out[idx] = acc[t][j] + x[idx];
            }
        }
    }
}

// ---------------------------------------------------------------------------
extern "C" void kernel_launch(void* const* d_in, const int* in_sizes, int n_in,
                              void* d_out, int out_size, void* d_ws, size_t ws_size,
                              hipStream_t stream) {
    const float* x  = (const float*)d_in[0];
    const float* Wq = (const float*)d_in[1];
    const float* bq = (const float*)d_in[2];
    const float* Wk = (const float*)d_in[3];
    const float* bk = (const float*)d_in[4];
    const float* Wv = (const float*)d_in[5];
    const float* bv = (const float*)d_in[6];
    float* out = (float*)d_out;

    __bf16* A   = (__bf16*)d_ws;
    __bf16* Wqb = A   + (size_t)32768 * 512;
    __bf16* Wkb = Wqb + (size_t)1024 * 512;
    __bf16* Wvb = Wkb + (size_t)1024 * 512;
    __bf16* Q   = Wvb + (size_t)1024 * 512;
    __bf16* K   = Q   + (size_t)32 * 1024 * 1024;
    __bf16* VT  = K   + (size_t)32 * 1024 * 1024;

    wiaa_haar_bf16 <<<65536, 256, 0, stream>>>(x, A);
    wiaa_wpack_bf16<<<2048,  256, 0, stream>>>(Wq, Wqb);
    wiaa_wpack_bf16<<<2048,  256, 0, stream>>>(Wk, Wkb);
    wiaa_wpack_bf16<<<2048,  256, 0, stream>>>(Wv, Wvb);
    wiaa_gemm_bf16 <<<4096,  256, 0, stream>>>(A, Wqb, bq, Q,  0);
    wiaa_gemm_bf16 <<<4096,  256, 0, stream>>>(A, Wkb, bk, K,  0);
    wiaa_gemm_bf16 <<<4096,  256, 0, stream>>>(A, Wvb, bv, VT, 1);
    wiaa_attn      <<<2048,  256, 0, stream>>>(Q, K, VT, x, out);
}
